// MSAColumnGlobalAttention_1279900254327
// MI455X (gfx1250) — matrix operand order repaired and checked by reading, not compiled, not run on hardware
//
#include <hip/hip_runtime.h>
#include <hip/hip_bf16.h>
#include <math.h>

typedef _Float16 half_t;
typedef __attribute__((ext_vector_type(16))) _Float16 v16h;
typedef __attribute__((ext_vector_type(8)))  float    v8f;

#define S_DIM 256
#define I_DIM 512
#define C_DIM 256
#define H_DIM 8
#define D_DIM 32
#define HD_DIM 256
#define NTOK (S_DIM * I_DIM)   // 131072 tokens

// ---------------------------------------------------------------------------
// WMMA fragment helpers (wave32, V_WMMA_F32_16X16X32_F16 layouts used here: A and B in contiguous k order)
// A (16x32 f16): lane L -> row M=L%16; element e holds K=e+(L/16)*16
// B (32x16 f16): lane L -> col N=L%16; element e holds K=e+(L/16)*16
// C/D (16x16 f32): VGPR r, lane L -> M=r+(L/16)*8, N=L%16
// Packed-B storage: Wp[(tile*32 + lane)*16 + e]  (32 contiguous bytes per lane)
// Packed-A storage: Ap[(tile*32 + lane)*16 + e]
// ---------------------------------------------------------------------------
__device__ __forceinline__ v8f wmma_f32_16x16x32(v16h a, v16h b, v8f c) {
  return __builtin_amdgcn_wmma_f32_16x16x32_f16(false, a, false, b, (short)0, c,
                                                false, false);
}

// A-fragment from row-major [M,K] tile (one contiguous 32B run per lane)
__device__ __forceinline__ v16h wmma_load_a(const half_t* base, int ld, int lane) {
  const half_t* p = base + (lane & 15) * ld + ((lane >> 4) << 4);
  v16h a;
#pragma unroll
  for (int e = 0; e < 16; ++e) {
    int k = e;
    a[e] = p[k];
  }
  return a;
}

// B[k][n] from a row-major [N=16, K] matrix (transpose-on-load): contiguous 32B
__device__ __forceinline__ v16h wmma_load_bt(const half_t* base, int ld, int lane) {
  const half_t* p = base + (lane & 15) * ld + ((lane >> 4) << 4);
  v16h b;
#pragma unroll
  for (int e = 0; e < 16; ++e) b[e] = p[e];
  return b;
}

// Packed fragment: single aligned 32-byte vector load
__device__ __forceinline__ v16h load_packed(const half_t* p, int tile, int lane) {
  return *(const v16h*)(p + (((size_t)tile << 5) + lane) * 16);
}

// ---------------------------------------------------------------------------
// 1) fp32 -> f16 weight conversion + swizzle into per-lane B-fragment layout.
//    src: row-major [256, N].  One thread per packed element.
// ---------------------------------------------------------------------------
__global__ void pack_b_kernel(const float* __restrict__ src,
                              half_t* __restrict__ dst, int N) {
  int idx = blockIdx.x * blockDim.x + threadIdx.x;   // total = 256*N
  int e = idx & 15;
  int lane = (idx >> 4) & 31;
  int tile = idx >> 9;                                // kt*(N/16) + nt
  int Ntiles = N >> 4;
  int kt = tile / Ntiles, nt = tile - kt * Ntiles;
  int k = kt * 32 + ((lane >> 4) << 4) + e;
  int n = nt * 16 + (lane & 15);
  dst[idx] = (half_t)src[k * N + n];
}

// ---------------------------------------------------------------------------
// 2) LayerNorm over C=256, one wave32 per token (8 tokens / 256-thread block)
// ---------------------------------------------------------------------------
__global__ void __launch_bounds__(256) ln_kernel(const float* __restrict__ x,
                                                 const float* __restrict__ gam,
                                                 const float* __restrict__ bet,
                                                 half_t* __restrict__ xln) {
  int token = blockIdx.x * 8 + (threadIdx.x >> 5);
  int lane = threadIdx.x & 31;
  const float* xr = x + (size_t)token * C_DIM;
  float v[8];
  float s = 0.f;
#pragma unroll
  for (int e = 0; e < 8; ++e) { v[e] = xr[lane + e * 32]; s += v[e]; }
#pragma unroll
  for (int off = 16; off > 0; off >>= 1) s += __shfl_xor(s, off, 32);
  float mu = s * (1.f / 256.f);
  float q = 0.f;
#pragma unroll
  for (int e = 0; e < 8; ++e) { float d = v[e] - mu; q += d * d; }
#pragma unroll
  for (int off = 16; off > 0; off >>= 1) q += __shfl_xor(q, off, 32);
  float inv = rsqrtf(q * (1.f / 256.f) + 1e-5f);
  half_t* o = xln + (size_t)token * C_DIM;
#pragma unroll
  for (int e = 0; e < 8; ++e) {
    int c = lane + e * 32;
    o[c] = (half_t)((v[e] - mu) * inv * gam[c] + bet[c]);
  }
}

// ---------------------------------------------------------------------------
// 3) xbar[i,c] = mean over s of x_ln[s,i,c]   (mean commutes with Wq)
// ---------------------------------------------------------------------------
__global__ void colmean_kernel(const half_t* __restrict__ xln,
                               half_t* __restrict__ xbar) {
  int idx = blockIdx.x * blockDim.x + threadIdx.x;  // i*C + c
  float s = 0.f;
  for (int t = 0; t < S_DIM; ++t)
    s += (float)xln[(size_t)t * (I_DIM * C_DIM) + idx];
  xbar[idx] = (half_t)(s * (1.f / (float)S_DIM));
}

// ---------------------------------------------------------------------------
// 4) Q = xbar[512,256] @ Wq[256,256], scaled by 1/sqrt(D).  512 wave-tiles.
// ---------------------------------------------------------------------------
__global__ void __launch_bounds__(256) gemm_q_kernel(const half_t* __restrict__ xbar,
                                                     const half_t* __restrict__ Wq_p,
                                                     half_t* __restrict__ Qs) {
  int w = blockIdx.x * 8 + (threadIdx.x >> 5);
  int lane = threadIdx.x & 31;
  int nt = w & 15;
  int mt = w >> 4;  // 0..31
  v8f acc = {};
#pragma unroll
  for (int kt = 0; kt < 8; ++kt) {
    v16h a = wmma_load_a(xbar + (size_t)mt * 16 * C_DIM + kt * 32, C_DIM, lane);
    v16h b = load_packed(Wq_p, kt * 16 + nt, lane);
    acc = wmma_f32_16x16x32(a, b, acc);
  }
  const float scale = 0.17677669529663689f;  // 1/sqrt(32)
  half_t* p = Qs + ((size_t)mt * 16 + ((lane >> 4) << 3)) * HD_DIM + nt * 16 + (lane & 15);
#pragma unroll
  for (int r = 0; r < 8; ++r) p[r * HD_DIM] = (half_t)(acc[r] * scale);
}

// ---------------------------------------------------------------------------
// 5) K,V = x_ln[131072,256] @ W[256,32].  32768 wave-tiles (2 mats x 8192 x 2).
// ---------------------------------------------------------------------------
__global__ void __launch_bounds__(256) gemm_kv_kernel(const half_t* __restrict__ xln,
                                                      const half_t* __restrict__ Wk_p,
                                                      const half_t* __restrict__ Wv_p,
                                                      half_t* __restrict__ Kh,
                                                      half_t* __restrict__ Vh) {
  int w = blockIdx.x * 8 + (threadIdx.x >> 5);
  int lane = threadIdx.x & 31;
  int nt = w & 1;
  int mt = (w >> 1) & 8191;
  int sel = (w >> 14) & 1;
  const half_t* W = sel ? Wv_p : Wk_p;
  half_t* O = sel ? Vh : Kh;
  const half_t* A = xln + (size_t)mt * 16 * C_DIM;
  v8f acc = {};
#pragma unroll
  for (int kt = 0; kt < 8; ++kt) {
    v16h a = wmma_load_a(A + kt * 32, C_DIM, lane);
    v16h b = load_packed(W, kt * 2 + nt, lane);
    acc = wmma_f32_16x16x32(a, b, acc);
  }
  half_t* p = O + ((size_t)mt * 16 + ((lane >> 4) << 3)) * D_DIM + nt * 16 + (lane & 15);
#pragma unroll
  for (int r = 0; r < 8; ++r) p[r * D_DIM] = (half_t)acc[r];
}

// ---------------------------------------------------------------------------
// 6) Per-column attention: scores = K_i @ Q_i^T, softmax over t, attn^T @ V_i
//    One block (8 waves) per residue i.  LDS: 57 KB.
//    V staged in packed-B layout; attn written in packed-A layout, so both
//    WMMA operand streams in the second GEMM are contiguous 32B LDS loads.
// ---------------------------------------------------------------------------
__global__ void __launch_bounds__(256) attn_kernel(const half_t* __restrict__ Qs,
                                                   const half_t* __restrict__ Kh,
                                                   const half_t* __restrict__ Vh,
                                                   half_t* __restrict__ weighted) {
  __shared__ __align__(32) half_t sK[S_DIM * D_DIM];   // 16 KB, row-major [t][d]
  __shared__ __align__(32) half_t sVp[16 * 512];       // 16 KB, packed-B (8 kt x 2 nt)
  __shared__ __align__(32) half_t sQ[16 * D_DIM];      // 1 KB (rows h, padded to 16)
  __shared__ __align__(32) float  sS[S_DIM * 16];      // 16 KB scores [t][h]
  __shared__ __align__(32) half_t sAp[8 * 512];        // 8 KB, packed-A (8 kt)
  int i = blockIdx.x;
  int t = threadIdx.x;
  {
    const uint4* ks = (const uint4*)(Kh + ((size_t)t * I_DIM + i) * D_DIM);
    uint4* kd = (uint4*)(sK + t * D_DIM);
#pragma unroll
    for (int j = 0; j < 4; ++j) kd[j] = ks[j];
    // V row -> registers -> packed-B LDS layout
    uint4 raw[4];
    const uint4* vs = (const uint4*)(Vh + ((size_t)t * I_DIM + i) * D_DIM);
#pragma unroll
    for (int j = 0; j < 4; ++j) raw[j] = vs[j];
    const half_t* vrow = (const half_t*)raw;
    int kt = t >> 5, khi = (t >> 4) & 1, e = t & 15;
#pragma unroll
    for (int d = 0; d < 32; ++d) {
      int tile = (kt << 1) | (d >> 4);
      int lane2 = (khi << 4) | (d & 15);
      sVp[(tile << 9) + (lane2 << 4) + e] = vrow[d];
    }
  }
  for (int idx = threadIdx.x; idx < 16 * D_DIM; idx += 256)
    sQ[idx] = (idx < HD_DIM) ? Qs[(size_t)i * HD_DIM + idx] : (half_t)0.0f;
  __syncthreads();

  int wid = threadIdx.x >> 5, lane = threadIdx.x & 31;
  // scores[t][h] : 16 t-tiles, each wave does 2 (1/sqrt(D) folded into Qs)
  for (int tt = wid * 2; tt < wid * 2 + 2; ++tt) {
    v16h a = wmma_load_a(sK + tt * 16 * D_DIM, D_DIM, lane);
    v16h b = wmma_load_bt(sQ, D_DIM, lane);  // B[k][n] = Q[h=n][d=k], contiguous
    v8f acc = {};
    acc = wmma_f32_16x16x32(a, b, acc);
    float* p = sS + tt * 16 * 16 + ((lane >> 4) << 3) * 16 + (lane & 15);
#pragma unroll
    for (int r = 0; r < 8; ++r) p[r * 16] = acc[r];
  }
  __syncthreads();

  // softmax over t (256) per head column; 16-lane group per column.
  // Writes attn directly in packed-A fragment layout.
  {
    int h = wid * 2 + (lane >> 4);
    int l = lane & 15;
    float vals[16];
    float m = -1e30f;
#pragma unroll
    for (int j = 0; j < 16; ++j) {
      float sv = sS[(j * 16 + l) * 16 + h];
      vals[j] = sv;
      m = fmaxf(m, sv);
    }
#pragma unroll
    for (int off = 8; off > 0; off >>= 1) m = fmaxf(m, __shfl_xor(m, off, 32));
    float sum = 0.f;
#pragma unroll
    for (int j = 0; j < 16; ++j) { float e = __expf(vals[j] - m); vals[j] = e; sum += e; }
#pragma unroll
    for (int off = 8; off > 0; off >>= 1) sum += __shfl_xor(sum, off, 32);
    float rs = 1.f / sum;
#pragma unroll
    for (int j = 0; j < 16; ++j) {
      int tt = j * 16 + l;                 // t index
      int k = tt & 31;
      int e = k & 15;
      int lane2 = (((k >> 4) & 1) << 4) | h;
      sAp[((tt >> 5) << 9) + (lane2 << 4) + e] = (half_t)(vals[j] * rs);
    }
  }
  __syncthreads();

  // weighted[h][d] = sum_t attn[t][h] * V[t][d] : waves 0/1 take d-tiles 0/1
  if (wid < 2) {
    v8f acc = {};
#pragma unroll
    for (int kt = 0; kt < 8; ++kt) {
      v16h a = load_packed(sAp, kt, lane);             // A = attn^T
      v16h b = load_packed(sVp, (kt << 1) | wid, lane);
      acc = wmma_f32_16x16x32(a, b, acc);
    }
    if (lane < 16) {  // rows M=0..7 == valid heads
      half_t* p = weighted + (size_t)i * HD_DIM + wid * 16 + lane;
#pragma unroll
      for (int r = 0; r < 8; ++r) p[r * D_DIM] = (half_t)acc[r];
    }
  }
}

// ---------------------------------------------------------------------------
// 7) Fused: G = sigmoid(x_ln @ Wg + bg); values = G * weighted[i]; out = values @ Wo + bo
//    16-token tile per block; values tile never leaves LDS.
// ---------------------------------------------------------------------------
__global__ void __launch_bounds__(256) fused_out_kernel(
    const half_t* __restrict__ xln, const half_t* __restrict__ Wg_p,
    const float* __restrict__ bg, const half_t* __restrict__ weighted,
    const half_t* __restrict__ Wo_p, const float* __restrict__ bo,
    float* __restrict__ out) {
  __shared__ __align__(32) half_t sX[16 * 256];    // 8 KB
  __shared__ __align__(32) half_t sVal[16 * 256];  // 8 KB
  int t0 = blockIdx.x * 16;        // 16 consecutive flat tokens (same s: 16 | I)
  int i0 = t0 & (I_DIM - 1);
  // gfx1250 global_prefetch_b8 on the (hot, reused) packed weights
  __builtin_prefetch(Wg_p + ((size_t)threadIdx.x << 6), 0, 1);
  __builtin_prefetch(Wo_p + ((size_t)threadIdx.x << 6), 0, 1);
  {
    const uint4* src = (const uint4*)(xln + (size_t)t0 * C_DIM);
    uint4* dst = (uint4*)sX;
    dst[threadIdx.x] = src[threadIdx.x];
    dst[threadIdx.x + 256] = src[threadIdx.x + 256];
  }
  __syncthreads();
  int wid = threadIdx.x >> 5, lane = threadIdx.x & 31;

  // stage 1: gated values tile -> LDS
  for (int nt = wid * 2; nt < wid * 2 + 2; ++nt) {
    v8f acc = {};
#pragma unroll
    for (int kt = 0; kt < 8; ++kt) {
      v16h a = wmma_load_a(sX + kt * 32, C_DIM, lane);
      v16h b = load_packed(Wg_p, kt * 16 + nt, lane);
      acc = wmma_f32_16x16x32(a, b, acc);
    }
    int col = nt * 16 + (lane & 15);
    float bgv = bg[col];
#pragma unroll
    for (int r = 0; r < 8; ++r) {
      int M = r + ((lane >> 4) << 3);
      float g = 1.f / (1.f + __expf(-(acc[r] + bgv)));
      float wv = (float)weighted[(size_t)(i0 + M) * HD_DIM + col];
      sVal[M * 256 + col] = (half_t)(g * wv);
    }
  }
  __syncthreads();

  // stage 2: out tile = values @ Wo + bo
  for (int nt = wid * 2; nt < wid * 2 + 2; ++nt) {
    v8f acc = {};
#pragma unroll
    for (int kt = 0; kt < 8; ++kt) {
      v16h a = wmma_load_a(sVal + kt * 32, 256, lane);
      v16h b = load_packed(Wo_p, kt * 16 + nt, lane);
      acc = wmma_f32_16x16x32(a, b, acc);
    }
    int col = nt * 16 + (lane & 15);
    float bov = bo[col];
    float* p = out + ((size_t)t0 + ((lane >> 4) << 3)) * C_DIM + col;
#pragma unroll
    for (int r = 0; r < 8; ++r) p[r * C_DIM] = acc[r] + bov;
  }
}

// ---------------------------------------------------------------------------
extern "C" void kernel_launch(void* const* d_in, const int* in_sizes, int n_in,
                              void* d_out, int out_size, void* d_ws, size_t ws_size,
                              hipStream_t stream) {
  (void)in_sizes; (void)n_in; (void)out_size; (void)ws_size;
  const float* x   = (const float*)d_in[0];
  const float* gam = (const float*)d_in[1];
  const float* bet = (const float*)d_in[2];
  const float* Wq  = (const float*)d_in[3];
  const float* Wk  = (const float*)d_in[4];
  const float* Wv  = (const float*)d_in[5];
  const float* Wg  = (const float*)d_in[6];
  const float* bg  = (const float*)d_in[7];
  const float* Wo  = (const float*)d_in[8];
  const float* bo  = (const float*)d_in[9];
  float* out = (float*)d_out;

  char* base = (char*)d_ws;
  size_t off = 0;
  auto carve = [&](size_t bytes) -> void* {
    void* p = base + off;
    off += (bytes + 255) & ~(size_t)255;
    return p;
  };
  half_t* xln  = (half_t*)carve((size_t)NTOK * C_DIM * sizeof(half_t));   // 64 MB
  half_t* xbar = (half_t*)carve((size_t)I_DIM * C_DIM * sizeof(half_t));
  half_t* Qs   = (half_t*)carve((size_t)I_DIM * HD_DIM * sizeof(half_t));
  half_t* Kh   = (half_t*)carve((size_t)NTOK * D_DIM * sizeof(half_t));   // 8 MB
  half_t* Vh   = (half_t*)carve((size_t)NTOK * D_DIM * sizeof(half_t));   // 8 MB
  half_t* Wt   = (half_t*)carve((size_t)I_DIM * HD_DIM * sizeof(half_t));
  half_t* Wq_p = (half_t*)carve(65536 * sizeof(half_t));
  half_t* Wk_p = (half_t*)carve(8192 * sizeof(half_t));
  half_t* Wv_p = (half_t*)carve(8192 * sizeof(half_t));
  half_t* Wg_p = (half_t*)carve(65536 * sizeof(half_t));
  half_t* Wo_p = (half_t*)carve(65536 * sizeof(half_t));

  pack_b_kernel<<<256, 256, 0, stream>>>(Wq, Wq_p, HD_DIM);
  pack_b_kernel<<<32, 256, 0, stream>>>(Wk, Wk_p, D_DIM);
  pack_b_kernel<<<32, 256, 0, stream>>>(Wv, Wv_p, D_DIM);
  pack_b_kernel<<<256, 256, 0, stream>>>(Wg, Wg_p, HD_DIM);
  pack_b_kernel<<<256, 256, 0, stream>>>(Wo, Wo_p, C_DIM);

  ln_kernel<<<NTOK / 8, 256, 0, stream>>>(x, gam, bet, xln);
  colmean_kernel<<<(I_DIM * C_DIM) / 256, 256, 0, stream>>>(xln, xbar);
  gemm_q_kernel<<<64, 256, 0, stream>>>(xbar, Wq_p, Qs);
  gemm_kv_kernel<<<4096, 256, 0, stream>>>(xln, Wk_p, Wv_p, Kh, Vh);
  attn_kernel<<<I_DIM, 256, 0, stream>>>(Qs, Kh, Vh, Wt);
  fused_out_kernel<<<NTOK / 16, 256, 0, stream>>>(xln, Wg_p, bg, Wt, Wo_p, bo, out);
}
